// GraphAwareTransformer_61435212202259
// MI455X (gfx1250) — matrix operand order repaired and checked by reading, not compiled, not run on hardware
//
#include <hip/hip_runtime.h>
#include <cstddef>
#include <cstdint>

// ---------------------------------------------------------------------------
// GraphAwareTransformer forward for MI455X (gfx1250, wave32, WMMA).
//   B=4096 items, N=64 neighbors, D=256, H=8 heads, Dh=32, OUT=256.
// Heavy GEMMs run on v_wmma_f32_16x16x32_f16 (f16 in, f32 accumulate).
// E-tile staging uses GLOBAL_LOAD_ASYNC_TO_LDS_B128 (ASYNCcnt-tracked DMA).
// ---------------------------------------------------------------------------

typedef __attribute__((ext_vector_type(16))) _Float16 v16h;
typedef __attribute__((ext_vector_type(8)))  float    v8f;

#define BATCH      4096
#define NNB        64
#define DM         256
#define NH         8
#define DH         32
#define SE_STRIDE  260   // f16 elements per LDS row for K/V (256 + pad)
#define SF_STRIDE  260   // f32 elements per LDS row for E   (256 + pad)

// ---------------- workspace layout (bytes) ----------------
//  wqf   : 0x00000  (128 KB)  Wq f16 B-fragments  (8 kt x 16 nt)
//  wkf   : 0x20000  (128 KB)
//  wvf   : 0x40000  (128 KB)
//  wof   : 0x60000  (128 KB)
//  wff   : 0x80000  (256 KB)  Wf (16 kt x 16 nt)
//  cdeA  : 0xC0000  (2 MB)    cde_hidden as f16 A-fragments (256 mtile x 8 kt)
//  attnA : cdeA+2MB (2 MB)    attention output as f16 A-fragments
//  qf32  : attnA+2MB (4 MB)   Q projection, f32 row-major [4096][256]

// B-fragment element:  frag[(kt*NT+nt)*32 + lane][e] = W[nt*16 + lane%16][kt*32 + 16*(e/8) + 8*(lane/16) + e%8]
__global__ void __launch_bounds__(256) gat_prep_wfrag(const float* __restrict__ W,
                                                      _Float16* __restrict__ dst, int KT) {
    int tid = blockIdx.x * 256 + threadIdx.x;      // one thread per lane-fragment
    if (tid >= KT * 16 * 32) return;
    int lane = tid & 31;
    int in_dim = KT * 32;
    int kt = tid >> 9;                              // 16 nt * 32 lanes = 512 per kt
    int o  = (((tid >> 5) & 15) << 4) + (lane & 15);
    int kb = kt * 32 + ((lane >> 4) << 3);
    #pragma unroll
    for (int e = 0; e < 16; ++e)
        dst[(size_t)tid * 16 + e] = (_Float16)W[(size_t)o * in_dim + kb + ((e & 8) << 1) + (e & 7)];
}

// A-fragment element: frag[(mtile*8+kt)*32 + lane][e] = X[mtile*16 + lane%16][kt*32 + 16*(e/8) + 8*(lane/16) + e%8]
__global__ void __launch_bounds__(256) gat_prep_cdea(const float* __restrict__ cde,
                                                     _Float16* __restrict__ dst) {
    int tid = blockIdx.x * 256 + threadIdx.x;       // 65536 threads total
    int lane = tid & 31;
    int kt = (tid >> 5) & 7;
    int mtile = tid >> 8;
    int m = lane & 15, g = lane >> 4;
    int row = mtile * 16 + m;
    #pragma unroll
    for (int e = 0; e < 16; ++e) {
        int k = kt * 32 + ((e & 8) << 1) + g * 8 + (e & 7);
        dst[(size_t)tid * 16 + e] = (_Float16)cde[(size_t)row * DM + k];
    }
}

// ---------------- Q projection: [4096 x 256] = cdeA @ Wq^T + bq ----------------
__global__ void __launch_bounds__(256) gat_qproj(const _Float16* __restrict__ cdeA,
                                                 const _Float16* __restrict__ wqf,
                                                 const float* __restrict__ bq,
                                                 float* __restrict__ qf32) {
    const int t = threadIdx.x, lane = t & 31, w = t >> 5;
    const int mtile = blockIdx.x;
    const int m = lane & 15, g = lane >> 4;
    v16h a[8];
    #pragma unroll
    for (int kt = 0; kt < 8; ++kt)
        a[kt] = *(const v16h*)(cdeA + (((size_t)mtile * 8 + kt) * 32 + lane) * 16);
    #pragma unroll
    for (int i = 0; i < 2; ++i) {
        int nt = w * 2 + i;
        v8f c = {};
        #pragma unroll
        for (int kt = 0; kt < 8; ++kt) {
            v16h bb = *(const v16h*)(wqf + (((size_t)kt * 16 + nt) * 32 + lane) * 16);
            c = __builtin_amdgcn_wmma_f32_16x16x32_f16(false, a[kt], false, bb,
                                                       (short)0, c, false, false);
        }
        int col = nt * 16 + m;
        float bi = bq[col];
        #pragma unroll
        for (int r = 0; r < 8; ++r)
            qf32[(size_t)(mtile * 16 + g * 8 + r) * DM + col] = c[r] + bi;
    }
}

// ---------------- fused attention: one workgroup per batch item ----------------
__global__ void __launch_bounds__(256) gat_attn(const float* __restrict__ nemb,
                                                const int*   __restrict__ etyp,
                                                const float* __restrict__ ebt,
                                                const float* __restrict__ bk,
                                                const float* __restrict__ bv,
                                                const float* __restrict__ qf32,
                                                const _Float16* __restrict__ wkf,
                                                const _Float16* __restrict__ wvf,
                                                _Float16* __restrict__ attnA) {
    __shared__ __align__(16) float    sEf32[NNB * SF_STRIDE]; // E tile, f32 (async DMA target)
    __shared__ __align__(16) _Float16 sK[NNB * SE_STRIDE];
    __shared__ __align__(16) _Float16 sV[NNB * SE_STRIDE];
    __shared__ __align__(16) float sQ[DM];
    __shared__ float sLog[NH * NNB];
    __shared__ float sBiasTab[64];
    __shared__ int   sEt[NNB];

    const int t = threadIdx.x, b = blockIdx.x;
    const int lane = t & 31, w = t >> 5;

    // ---- async DMA: E tile [64 x 256] f32, HBM -> LDS, no VGPR round-trip ----
    // GVS mode: vdst = LDS byte offset, vaddr = 32-bit byte offset, saddr = base.
    {
        const char* ebase = (const char*)(nemb + (size_t)b * NNB * DM);
        #pragma unroll 4
        for (int i = 0; i < 16; ++i) {
            int lin = i * 256 + t;                  // 16-byte chunk index (4096 total)
            int row = lin >> 6, cv = lin & 63;      // 64 chunks per 256-float row
            unsigned goff = (unsigned)lin * 16u;
            unsigned loff = (unsigned)(uintptr_t)&sEf32[row * SF_STRIDE + cv * 4];
            asm volatile("global_load_async_to_lds_b128 %0, %1, %2"
                         :: "v"(loff), "v"(goff), "s"(ebase) : "memory");
        }
    }
    sQ[t] = qf32[(size_t)b * DM + t];
    if (t < NNB) { sEt[t] = etyp[b * NNB + t]; sBiasTab[t] = ebt[t]; }
    asm volatile("s_wait_asynccnt 0x0" ::: "memory");
    __syncthreads();

    // K = E@Wk^T + bk, V = E@Wv^T + bv via WMMA.
    // wave -> (matrix, m-tile pair, n-half); 2 M-tiles share each B load.
    {
        const int mat   = (w >> 2) & 1;             // 0: K, 1: V
        const int mpair = (w >> 1) & 1;             // m-tiles {0,1} or {2,3}
        const int nh    = w & 1;
        const int m = lane & 15, g = lane >> 4;
        v16h a0[8], a1[8];
        const int r0 = mpair * 32 + m, r1 = r0 + 16;
        const float* er0 = &sEf32[r0 * SF_STRIDE];
        const float* er1 = &sEf32[r1 * SF_STRIDE];
        #pragma unroll
        for (int kt = 0; kt < 8; ++kt) {
            #pragma unroll
            for (int p = 0; p < 8; ++p) {
                int e0 = 2 * p;
                int K = kt * 32 + ((e0 & 8) << 1) + g * 8 + (e0 & 7);
                float2 f0 = *(const float2*)(er0 + K);
                float2 f1 = *(const float2*)(er1 + K);
                a0[kt][e0]     = (_Float16)f0.x;
                a0[kt][e0 + 1] = (_Float16)f0.y;
                a1[kt][e0]     = (_Float16)f1.x;
                a1[kt][e0 + 1] = (_Float16)f1.y;
            }
        }
        const _Float16* wf  = mat ? wvf : wkf;
        const float*    bia = mat ? bv : bk;
        _Float16*       dst = mat ? sV : sK;
        for (int i = 0; i < 8; ++i) {
            int nt = nh * 8 + i;
            v8f c0 = {}, c1 = {};
            #pragma unroll
            for (int kt = 0; kt < 8; ++kt) {
                v16h bb = *(const v16h*)(wf + (((size_t)kt * 16 + nt) * 32 + lane) * 16);
                c0 = __builtin_amdgcn_wmma_f32_16x16x32_f16(false, a0[kt], false, bb,
                                                            (short)0, c0, false, false);
                c1 = __builtin_amdgcn_wmma_f32_16x16x32_f16(false, a1[kt], false, bb,
                                                            (short)0, c1, false, false);
            }
            int col = nt * 16 + m;
            float bi = bia[col];
            #pragma unroll
            for (int r = 0; r < 8; ++r) {
                int rowa = mpair * 32 + g * 8 + r;
                dst[rowa * SE_STRIDE + col]        = (_Float16)(c0[r] + bi);
                dst[(rowa + 16) * SE_STRIDE + col] = (_Float16)(c1[r] + bi);
            }
        }
    }
    __syncthreads();

    // logits[h][n] = scale * <Qh, Kh[n]> + edge_bias[etype[n]][h]
    const float scale = 0.17677669529663687f;       // 1/sqrt(32)
    #pragma unroll
    for (int pp = 0; pp < 2; ++pp) {
        int idx = t + pp * 256;
        int h = idx >> 6, n = idx & 63;
        float acc = 0.f;
        #pragma unroll 8
        for (int d = 0; d < DH; ++d)
            acc += sQ[h * DH + d] * (float)sK[n * SE_STRIDE + h * DH + d];
        sLog[idx] = acc * scale + sBiasTab[sEt[n] * NH + h];
    }
    __syncthreads();

    // softmax over n=64: wave w handles head w (2 values per lane, shfl_xor wave32)
    {
        float v0 = sLog[w * 64 + lane];
        float v1 = sLog[w * 64 + 32 + lane];
        float mx = fmaxf(v0, v1);
        #pragma unroll
        for (int o = 16; o >= 1; o >>= 1) mx = fmaxf(mx, __shfl_xor(mx, o, 32));
        float e0 = __expf(v0 - mx), e1 = __expf(v1 - mx);
        float s = e0 + e1;
        #pragma unroll
        for (int o = 16; o >= 1; o >>= 1) s += __shfl_xor(s, o, 32);
        float inv = 1.f / s;
        sLog[w * 64 + lane]      = e0 * inv;
        sLog[w * 64 + 32 + lane] = e1 * inv;
    }
    __syncthreads();

    // attn_out[t] = sum_n attn[h][n] * V[n][t]; scatter into f16 A-fragment layout
    {
        int h = t >> 5;
        float acc = 0.f;
        #pragma unroll 8
        for (int n = 0; n < NNB; ++n)
            acc += sLog[h * 64 + n] * (float)sV[n * SE_STRIDE + t];
        int mtile = b >> 4, m = b & 15;
        int kt = t >> 5, rem = t & 31;
        int e = ((rem >> 4) & 1) * 8 + (rem & 7);
        int lane2 = ((rem >> 3) & 1) * 16 + m;
        attnA[(((size_t)mtile * 8 + kt) * 32 + lane2) * 16 + e] = (_Float16)acc;
    }
}

// ---------------- output: O=attn@Wo^T+bo ; y=[O,cde]@Wf^T+bf ; LN ; ReLU ----------------
__global__ void __launch_bounds__(256) gat_out(const _Float16* __restrict__ attnA,
                                               const _Float16* __restrict__ cdeA,
                                               const _Float16* __restrict__ wof,
                                               const _Float16* __restrict__ wff,
                                               const float* __restrict__ bo,
                                               const float* __restrict__ bfv,
                                               const float* __restrict__ gam,
                                               const float* __restrict__ bet,
                                               float* __restrict__ out) {
    __shared__ __align__(32) _Float16 sOA[8 * 32 * 16];  // O as A-fragments
    __shared__ float sY[16 * SE_STRIDE];
    __shared__ float sSum[256], sSqs[256];
    __shared__ float sMu[16], sRs[16];

    const int t = threadIdx.x, lane = t & 31, w = t >> 5;
    const int mtile = blockIdx.x;
    const int m = lane & 15, g = lane >> 4;

    v16h a[8];
    #pragma unroll
    for (int kt = 0; kt < 8; ++kt)
        a[kt] = *(const v16h*)(attnA + (((size_t)mtile * 8 + kt) * 32 + lane) * 16);

    #pragma unroll
    for (int i = 0; i < 2; ++i) {
        int nt = w * 2 + i;
        v8f c = {};
        #pragma unroll
        for (int kt = 0; kt < 8; ++kt) {
            v16h bb = *(const v16h*)(wof + (((size_t)kt * 16 + nt) * 32 + lane) * 16);
            c = __builtin_amdgcn_wmma_f32_16x16x32_f16(false, a[kt], false, bb,
                                                       (short)0, c, false, false);
        }
        int col = nt * 16 + m;
        float bi = bo[col];
        int ktA = col >> 5, rem = col & 31;
        int e = ((rem >> 4) & 1) * 8 + (rem & 7);
        int gA = (rem >> 3) & 1;
        #pragma unroll
        for (int r = 0; r < 8; ++r) {
            int laneA = gA * 16 + (g * 8 + r);
            sOA[(ktA * 32 + laneA) * 16 + e] = (_Float16)(c[r] + bi);
        }
    }
    __syncthreads();

    #pragma unroll
    for (int i = 0; i < 2; ++i) {
        int nt = w * 2 + i;
        v8f c = {};
        #pragma unroll
        for (int kt = 0; kt < 16; ++kt) {
            v16h aa = (kt < 8)
                ? *(const v16h*)(sOA + ((size_t)kt * 32 + lane) * 16)
                : *(const v16h*)(cdeA + (((size_t)mtile * 8 + (kt - 8)) * 32 + lane) * 16);
            v16h bb = *(const v16h*)(wff + (((size_t)kt * 16 + nt) * 32 + lane) * 16);
            c = __builtin_amdgcn_wmma_f32_16x16x32_f16(false, aa, false, bb,
                                                       (short)0, c, false, false);
        }
        int col = nt * 16 + m;
        float bi = bfv[col];
        #pragma unroll
        for (int r = 0; r < 8; ++r)
            sY[(g * 8 + r) * SE_STRIDE + col] = c[r] + bi;
    }
    __syncthreads();

    // LayerNorm over 256 cols per row (16 rows in this tile) + ReLU
    {
        int r = t >> 4, ch = t & 15;
        float s = 0.f, q = 0.f;
        #pragma unroll
        for (int j = 0; j < 16; ++j) {
            float v = sY[r * SE_STRIDE + ch * 16 + j];
            s += v; q += v * v;
        }
        sSum[t] = s; sSqs[t] = q;
    }
    __syncthreads();
    if (t < 16) {
        float s = 0.f, q = 0.f;
        #pragma unroll
        for (int j = 0; j < 16; ++j) { s += sSum[t * 16 + j]; q += sSqs[t * 16 + j]; }
        float mu = s * (1.f / 256.f);
        float var = q * (1.f / 256.f) - mu * mu;
        sMu[t] = mu;
        sRs[t] = rsqrtf(var + 1e-5f);
    }
    __syncthreads();
    {
        int r = t >> 4, ch = t & 15;
        float mu = sMu[r], rs = sRs[r];
        #pragma unroll
        for (int j = 0; j < 16; ++j) {
            int col = ch * 16 + j;
            float v = (sY[r * SE_STRIDE + col] - mu) * rs * gam[col] + bet[col];
            out[(size_t)(mtile * 16 + r) * DM + col] = fmaxf(v, 0.f);
        }
    }
}

// ---------------------------------------------------------------------------
extern "C" void kernel_launch(void* const* d_in, const int* in_sizes, int n_in,
                              void* d_out, int out_size, void* d_ws, size_t ws_size,
                              hipStream_t stream) {
    const float* cde  = (const float*)d_in[0];
    const float* nemb = (const float*)d_in[1];
    const int*   et   = (const int*)  d_in[2];
    const float* Wq = (const float*)d_in[3];  const float* bq  = (const float*)d_in[4];
    const float* Wk = (const float*)d_in[5];  const float* bk  = (const float*)d_in[6];
    const float* Wv = (const float*)d_in[7];  const float* bv  = (const float*)d_in[8];
    const float* Wo = (const float*)d_in[9];  const float* bo  = (const float*)d_in[10];
    const float* ebt = (const float*)d_in[11];
    const float* Wf = (const float*)d_in[12]; const float* bfv = (const float*)d_in[13];
    const float* gam = (const float*)d_in[14]; const float* bet = (const float*)d_in[15];

    char* ws = (char*)d_ws;
    _Float16* wqf   = (_Float16*)(ws + 0x00000);
    _Float16* wkf   = (_Float16*)(ws + 0x20000);
    _Float16* wvf   = (_Float16*)(ws + 0x40000);
    _Float16* wof   = (_Float16*)(ws + 0x60000);
    _Float16* wff   = (_Float16*)(ws + 0x80000);
    _Float16* cdeA  = (_Float16*)(ws + 0xC0000);
    _Float16* attnA = (_Float16*)(ws + 0xC0000 + (1u << 21));
    float*    qf32  = (float*)   (ws + 0xC0000 + 2u * (1u << 21));
    // total workspace used: 0xC0000 + 2MB + 2MB + 4MB = ~8.75 MB

    gat_prep_wfrag<<<16, 256, 0, stream>>>(Wq, wqf, 8);
    gat_prep_wfrag<<<16, 256, 0, stream>>>(Wk, wkf, 8);
    gat_prep_wfrag<<<16, 256, 0, stream>>>(Wv, wvf, 8);
    gat_prep_wfrag<<<16, 256, 0, stream>>>(Wo, wof, 8);
    gat_prep_wfrag<<<32, 256, 0, stream>>>(Wf, wff, 16);
    gat_prep_cdea<<<256, 256, 0, stream>>>(cde, cdeA);

    gat_qproj<<<BATCH / 16, 256, 0, stream>>>(cdeA, wqf, bq, qf32);
    gat_attn<<<BATCH, 256, 0, stream>>>(nemb, et, ebt, bk, bv, qf32, wkf, wvf, attnA);
    gat_out<<<BATCH / 16, 256, 0, stream>>>(attnA, cdeA, wof, wff, bo, bfv, gam, bet,
                                            (float*)d_out);
}
